// MultiHeadAttention_18562848653415
// MI455X (gfx1250) — matrix operand order repaired and checked by reading, not compiled, not run on hardware
//
#include <hip/hip_runtime.h>

// MHA: x[4,2048,1024] @ w_qkv -> flash attention (16 heads, d=64) -> @ w_out + b.
// All GEMMs run on v_wmma_f32_16x16x32_bf16 (f32 accumulate).
// Byte-copy tile staging uses GLOBAL_LOAD_ASYNC_TO_LDS_B128 (ASYNCcnt) when the
// builtin exists; falls back to manual LDS stores otherwise.
// Workspace: Q,K,V [B,H,T,D] bf16 + O [B,T,H*D] bf16 = 64 MB.

typedef __bf16 bf16_t;
typedef __attribute__((ext_vector_type(8)))  __bf16 v8bf;
typedef __attribute__((ext_vector_type(16))) __bf16 v16bf;
typedef __attribute__((ext_vector_type(8)))  float  v8f;
typedef __attribute__((ext_vector_type(4)))  float  f32x4;
typedef __attribute__((ext_vector_type(4)))  int    v4i;

#ifndef __has_builtin
#define __has_builtin(x) 0
#endif

#if __has_builtin(__builtin_amdgcn_global_load_async_to_lds_b128)
#define HAVE_ASYNC_LDS 1
#else
#define HAVE_ASYNC_LDS 0
#endif

// Address-space-qualified pointer types matching the async builtin signature:
// (v4i addrspace(1)*, v4i addrspace(3)*, Ii offset, Ii cpol)
typedef __attribute__((address_space(1))) v4i gv4i;
typedef __attribute__((address_space(3))) v4i lv4i;

constexpr int B_  = 4;
constexpr int T_  = 2048;
constexpr int IN_ = 1024;   // input dim
constexpr int H_  = 16;     // heads
constexpr int D_  = 64;     // head dim
constexpr int HD_ = H_ * D_;        // 1024
constexpr int QKV_ = 3 * HD_;       // 3072
constexpr int M_  = B_ * T_;        // 8192 tokens

static __device__ inline v16bf mk16(v8bf lo, v8bf hi) {
    v16bf r;
#pragma unroll
    for (int i = 0; i < 8; ++i) { r[i] = lo[i]; r[i + 8] = hi[i]; }
    return r;
}
static __device__ inline v8f zero8() {
    v8f z;
#pragma unroll
    for (int i = 0; i < 8; ++i) z[i] = 0.0f;
    return z;
}
static __device__ inline v8f wmma_bf16(v16bf a, v16bf b, v8f c) {
    return __builtin_amdgcn_wmma_f32_16x16x32_bf16(false, a, false, b, (short)0, c,
                                                   false, false);
}

// 16-byte global -> LDS copy (async path when available).
static __device__ inline void g2l_16B(const bf16_t* g, bf16_t* l) {
#if HAVE_ASYNC_LDS
    __builtin_amdgcn_global_load_async_to_lds_b128((gv4i*)g, (lv4i*)l, 0, 0);
#else
    *(v8bf*)l = *(const v8bf*)g;
#endif
}
static __device__ inline void async_fence() {
#if HAVE_ASYNC_LDS
#if __has_builtin(__builtin_amdgcn_s_wait_asynccnt)
    __builtin_amdgcn_s_wait_asynccnt(0);
#else
    asm volatile("s_wait_asynccnt 0x0" ::: "memory");
#endif
#endif
}

// ---------------------------------------------------------------------------
// Kernel 1: qkv = x @ w_qkv, epilogue scatters bf16 into Q/K/V [B,H,T,D].
// Block tile 128(M) x 128(N), BK=32, 256 threads = 8 waves, wave tile 32x64.
// ---------------------------------------------------------------------------
__global__ void mha_qkv_gemm(const float* __restrict__ x,
                             const float* __restrict__ w,
                             bf16_t* __restrict__ Q,
                             bf16_t* __restrict__ K,
                             bf16_t* __restrict__ V) {
    __shared__ __attribute__((aligned(16))) bf16_t As[128][40]; // [m][k]
    __shared__ __attribute__((aligned(16))) bf16_t Bs[128][40]; // [n][k] (transposed)

    const int tid  = threadIdx.x;
    const int wave = tid >> 5;
    const int lane = tid & 31;
    const int m0   = blockIdx.y * 128;
    const int n0   = blockIdx.x * 128;
    const int wm   = (wave >> 1) * 32;     // wave M offset within block
    const int wn   = (wave & 1) * 64;      // wave N offset within block
    const int lm   = lane & 15;
    const int lg   = lane >> 4;            // lane group (0/1)

    v8f acc[2][4];
#pragma unroll
    for (int i = 0; i < 2; ++i)
#pragma unroll
        for (int j = 0; j < 4; ++j) acc[i][j] = zero8();

    for (int k0 = 0; k0 < IN_; k0 += 32) {
        // Stage A tile 128x32 (f32 -> bf16)
#pragma unroll
        for (int i = 0; i < 4; ++i) {
            int f  = tid + 256 * i;           // 0..1023 float4 chunks
            int r  = f >> 3;
            int c4 = (f & 7) << 2;
            f32x4 v = *(const f32x4*)(x + (size_t)(m0 + r) * IN_ + k0 + c4);
#pragma unroll
            for (int j = 0; j < 4; ++j) As[r][c4 + j] = (bf16_t)v[j];
        }
        // Stage B tile 32x128 transposed into Bs[n][k]
#pragma unroll
        for (int i = 0; i < 4; ++i) {
            int f  = tid + 256 * i;           // 0..1023 float4 chunks
            int r  = f >> 5;                  // k row 0..31
            int c4 = (f & 31) << 2;           // n col
            f32x4 v = *(const f32x4*)(w + (size_t)(k0 + r) * QKV_ + n0 + c4);
#pragma unroll
            for (int j = 0; j < 4; ++j) Bs[c4 + j][r] = (bf16_t)v[j];
        }
        __syncthreads();

        v16bf afr[2], bfr[4];
#pragma unroll
        for (int mt = 0; mt < 2; ++mt) {
            int r = wm + mt * 16 + lm;
            afr[mt] = mk16(*(const v8bf*)&As[r][lg * 8],
                           *(const v8bf*)&As[r][16 + lg * 8]);
        }
#pragma unroll
        for (int nt = 0; nt < 4; ++nt) {
            int r = wn + nt * 16 + lm;
            bfr[nt] = mk16(*(const v8bf*)&Bs[r][lg * 8],
                           *(const v8bf*)&Bs[r][16 + lg * 8]);
        }
#pragma unroll
        for (int mt = 0; mt < 2; ++mt)
#pragma unroll
            for (int nt = 0; nt < 4; ++nt)
                acc[mt][nt] = wmma_bf16(afr[mt], bfr[nt], acc[mt][nt]);
        __syncthreads();
    }

    // Epilogue: scatter into Q/K/V [B,H,T,D] (fused 'b t (qkv h d) -> qkv b h t d')
#pragma unroll
    for (int mt = 0; mt < 2; ++mt)
#pragma unroll
        for (int nt = 0; nt < 4; ++nt)
#pragma unroll
            for (int r = 0; r < 8; ++r) {
                int row = m0 + wm + mt * 16 + lg * 8 + r;   // token index
                int col = n0 + wn + nt * 16 + lm;           // qkv column
                int which = col >> 10;
                int rem   = col & 1023;
                int h     = rem >> 6;
                int d     = rem & 63;
                int b     = row >> 11;
                int t     = row & (T_ - 1);
                bf16_t* dst = (which == 0) ? Q : (which == 1) ? K : V;
                dst[(((size_t)(b * H_ + h)) * T_ + t) * D_ + d] =
                    (bf16_t)acc[mt][nt][r];
            }
}

// ---------------------------------------------------------------------------
// Kernel 2: flash attention. One block = one (b,h) and 64 query rows.
// S = Q K^T (WMMA), online softmax through LDS (4 threads/row),
// O += P V (WMMA, C from LDS). Q/K tiles staged with async global->LDS.
// ---------------------------------------------------------------------------
__global__ void mha_flash_attn(const bf16_t* __restrict__ Q,
                               const bf16_t* __restrict__ K,
                               const bf16_t* __restrict__ V,
                               bf16_t* __restrict__ O) {
    __shared__ __attribute__((aligned(16))) bf16_t Qs[64][72]; // [q][d]
    __shared__ __attribute__((aligned(16))) bf16_t Ks[64][72]; // [key][d]
    __shared__ __attribute__((aligned(16))) bf16_t Vs[64][72]; // [d][key] (transposed)
    __shared__ __attribute__((aligned(16))) bf16_t Pb[64][72]; // [q][key]
    __shared__ float Sf[64][66];
    __shared__ float Of[64][66];
    __shared__ float pmax[64][4];   // partial row max (4 segments/row)
    __shared__ float psum[64][4];   // partial row sum
    __shared__ float m_run[64], l_run[64], arow[64], mrow[64];

    const int tid  = threadIdx.x;
    const int wave = tid >> 5;
    const int lane = tid & 31;
    const int lm   = lane & 15;
    const int lg   = lane >> 4;
    const int bh   = blockIdx.y;
    const int bb   = bh >> 4;
    const int hh   = bh & 15;
    const int m0   = blockIdx.x * 64;
    const float scale = 0.125f; // 1/sqrt(64)

    const bf16_t* Qh = Q + ((size_t)(bb * H_ + hh)) * T_ * D_;
    const bf16_t* Kh = K + ((size_t)(bb * H_ + hh)) * T_ * D_;
    const bf16_t* Vh = V + ((size_t)(bb * H_ + hh)) * T_ * D_;

    // softmax work split: 4 threads per row, 16 cols per thread
    const int srow = tid >> 2;
    const int sseg = tid & 3;
    const int sc0  = sseg * 16;

    // Init accumulators / stats; load Q tile once (async copy).
    for (int idx = tid; idx < 64 * 64; idx += 256) Of[idx >> 6][idx & 63] = 0.0f;
    if (tid < 64) { m_run[tid] = -1e30f; l_run[tid] = 0.0f; }
#pragma unroll
    for (int i = 0; i < 2; ++i) {
        int f  = tid + 256 * i;          // 512 chunks of 8 bf16
        int r  = f >> 3;
        int c8 = (f & 7) << 3;
        g2l_16B(Qh + (size_t)(m0 + r) * D_ + c8, &Qs[r][c8]);
    }
    async_fence();
    __syncthreads();

    for (int kt = 0; kt < T_ / 64; ++kt) {
        const int kb = kt * 64;
        // Stage K [key][d] (async) and V transposed [d][key] (register transpose)
#pragma unroll
        for (int i = 0; i < 2; ++i) {
            int f  = tid + 256 * i;
            int r  = f >> 3;
            int c8 = (f & 7) << 3;
            g2l_16B(Kh + (size_t)(kb + r) * D_ + c8, &Ks[r][c8]);
            v8bf v = *(const v8bf*)(Vh + (size_t)(kb + r) * D_ + c8);
#pragma unroll
            for (int j = 0; j < 8; ++j) Vs[c8 + j][r] = v[j];
        }
        async_fence();
        __syncthreads();

        // S = Q K^T : 16 tiles of 16x16, 2 per wave.
#pragma unroll
        for (int i = 0; i < 2; ++i) {
            int tile = wave * 2 + i;
            int mt = tile >> 2, nt = tile & 3;
            v8f acc = zero8();
#pragma unroll
            for (int kc = 0; kc < 64; kc += 32) {
                int qr = mt * 16 + lm;
                v16bf a = mk16(*(const v8bf*)&Qs[qr][kc + lg * 8],
                               *(const v8bf*)&Qs[qr][kc + 16 + lg * 8]);
                int kr = nt * 16 + lm;
                v16bf b = mk16(*(const v8bf*)&Ks[kr][kc + lg * 8],
                               *(const v8bf*)&Ks[kr][kc + 16 + lg * 8]);
                acc = wmma_bf16(a, b, acc);
            }
#pragma unroll
            for (int r = 0; r < 8; ++r)
                Sf[mt * 16 + lg * 8 + r][nt * 16 + lm] = acc[r] * scale;
        }
        __syncthreads();

        // Online softmax, parallel: phase 1 - partial max per 16-col segment.
        {
            float mx = -1e30f;
#pragma unroll
            for (int c = 0; c < 16; ++c) mx = fmaxf(mx, Sf[srow][sc0 + c]);
            pmax[srow][sseg] = mx;
        }
        __syncthreads();
        // phase 2 - row max / alpha (one thread per row)
        if (tid < 64) {
            int r = tid;
            float mo = m_run[r];
            float mx = fmaxf(fmaxf(pmax[r][0], pmax[r][1]),
                             fmaxf(pmax[r][2], pmax[r][3]));
            mx = fmaxf(mo, mx);
            arow[r] = expf(mo - mx);
            mrow[r] = mx;
            m_run[r] = mx;
        }
        __syncthreads();
        // phase 3 - exp, P write, partial sums, O rescale (4 threads/row)
        {
            float mx = mrow[srow];
            float al = arow[srow];
            float s = 0.0f;
#pragma unroll
            for (int c = 0; c < 16; ++c) {
                float p = expf(Sf[srow][sc0 + c] - mx);
                Pb[srow][sc0 + c] = (bf16_t)p;
                s += p;
                Of[srow][sc0 + c] *= al;
            }
            psum[srow][sseg] = s;
        }
        __syncthreads();
        // phase 4 - fold partial sums into running l
        if (tid < 64) {
            int r = tid;
            l_run[r] = l_run[r] * arow[r] +
                       (psum[r][0] + psum[r][1]) + (psum[r][2] + psum[r][3]);
        }
        __syncthreads();

        // O += P @ V : 16 tiles of 16x16, 2 per wave, C sourced from LDS.
#pragma unroll
        for (int i = 0; i < 2; ++i) {
            int tile = wave * 2 + i;
            int mt = tile >> 2, nt = tile & 3;
            v8f acc;
#pragma unroll
            for (int r = 0; r < 8; ++r)
                acc[r] = Of[mt * 16 + lg * 8 + r][nt * 16 + lm];
#pragma unroll
            for (int kc = 0; kc < 64; kc += 32) {
                int qr = mt * 16 + lm;
                v16bf a = mk16(*(const v8bf*)&Pb[qr][kc + lg * 8],
                               *(const v8bf*)&Pb[qr][kc + 16 + lg * 8]);
                int dr = nt * 16 + lm;
                v16bf b = mk16(*(const v8bf*)&Vs[dr][kc + lg * 8],
                               *(const v8bf*)&Vs[dr][kc + 16 + lg * 8]);
                acc = wmma_bf16(a, b, acc);
            }
#pragma unroll
            for (int r = 0; r < 8; ++r)
                Of[mt * 16 + lg * 8 + r][nt * 16 + lm] = acc[r];
        }
        __syncthreads();
    }

    // Finalize: O / l, write bf16 to [B, T, H*D] (fuses head-transpose).
    for (int idx = tid; idx < 64 * 64; idx += 256) {
        int r = idx >> 6, d = idx & 63;
        float val = Of[r][d] / l_run[r];
        int t = m0 + r;
        O[((size_t)(bb * T_ + t)) * HD_ + hh * D_ + d] = (bf16_t)val;
    }
}

// ---------------------------------------------------------------------------
// Kernel 3: out = O @ w_out + b_out (f32 output).
// Block tile 128x128, BK=32, wave tile 32x64. A staged via async copy.
// ---------------------------------------------------------------------------
__global__ void mha_out_gemm(const bf16_t* __restrict__ A,   // [8192, 1024] bf16
                             const float* __restrict__ w,    // [1024, 1024]
                             const float* __restrict__ bias, // [1024]
                             float* __restrict__ out) {
    __shared__ __attribute__((aligned(16))) bf16_t As[128][40];
    __shared__ __attribute__((aligned(16))) bf16_t Bs[128][40]; // [n][k]

    const int tid  = threadIdx.x;
    const int wave = tid >> 5;
    const int lane = tid & 31;
    const int m0   = blockIdx.y * 128;
    const int n0   = blockIdx.x * 128;
    const int wm   = (wave >> 1) * 32;
    const int wn   = (wave & 1) * 64;
    const int lm   = lane & 15;
    const int lg   = lane >> 4;

    v8f acc[2][4];
#pragma unroll
    for (int i = 0; i < 2; ++i)
#pragma unroll
        for (int j = 0; j < 4; ++j) acc[i][j] = zero8();

    for (int k0 = 0; k0 < HD_; k0 += 32) {
        // A tile 128x32 bf16 (byte copy -> async path)
#pragma unroll
        for (int i = 0; i < 2; ++i) {
            int f  = tid + 256 * i;   // 512 chunks of 8 bf16
            int r  = f >> 2;
            int c8 = (f & 3) << 3;
            g2l_16B(A + (size_t)(m0 + r) * HD_ + k0 + c8, &As[r][c8]);
        }
        // B tile 32x128 f32 -> bf16 transposed
#pragma unroll
        for (int i = 0; i < 4; ++i) {
            int f  = tid + 256 * i;
            int r  = f >> 5;
            int c4 = (f & 31) << 2;
            f32x4 v = *(const f32x4*)(w + (size_t)(k0 + r) * IN_ + n0 + c4);
#pragma unroll
            for (int j = 0; j < 4; ++j) Bs[c4 + j][r] = (bf16_t)v[j];
        }
        async_fence();
        __syncthreads();

        v16bf afr[2], bfr[4];
#pragma unroll
        for (int mt = 0; mt < 2; ++mt) {
            int r = wm + mt * 16 + lm;
            afr[mt] = mk16(*(const v8bf*)&As[r][lg * 8],
                           *(const v8bf*)&As[r][16 + lg * 8]);
        }
#pragma unroll
        for (int nt = 0; nt < 4; ++nt) {
            int r = wn + nt * 16 + lm;
            bfr[nt] = mk16(*(const v8bf*)&Bs[r][lg * 8],
                           *(const v8bf*)&Bs[r][16 + lg * 8]);
        }
#pragma unroll
        for (int mt = 0; mt < 2; ++mt)
#pragma unroll
            for (int nt = 0; nt < 4; ++nt)
                acc[mt][nt] = wmma_bf16(afr[mt], bfr[nt], acc[mt][nt]);
        __syncthreads();
    }

#pragma unroll
    for (int mt = 0; mt < 2; ++mt)
#pragma unroll
        for (int nt = 0; nt < 4; ++nt)
#pragma unroll
            for (int r = 0; r < 8; ++r) {
                int row = m0 + wm + mt * 16 + lg * 8 + r;
                int col = n0 + wn + nt * 16 + lm;
                out[(size_t)row * IN_ + col] = acc[mt][nt][r] + bias[col];
            }
}

// ---------------------------------------------------------------------------
extern "C" void kernel_launch(void* const* d_in, const int* in_sizes, int n_in,
                              void* d_out, int out_size, void* d_ws, size_t ws_size,
                              hipStream_t stream) {
    (void)in_sizes; (void)n_in; (void)out_size; (void)ws_size;
    const float* x     = (const float*)d_in[0];
    const float* w_qkv = (const float*)d_in[1];
    const float* w_out = (const float*)d_in[2];
    const float* b_out = (const float*)d_in[3];
    float* out = (float*)d_out;

    const size_t headElems = (size_t)B_ * H_ * T_ * D_;  // 8,388,608 per tensor
    bf16_t* Q = (bf16_t*)d_ws;
    bf16_t* K = Q + headElems;
    bf16_t* V = K + headElems;
    bf16_t* O = V + headElems;   // [B, T, H*D] bf16; total ws use = 64 MB

    mha_qkv_gemm<<<dim3(QKV_ / 128, M_ / 128), 256, 0, stream>>>(x, w_qkv, Q, K, V);
    mha_flash_attn<<<dim3(T_ / 64, B_ * H_), 256, 0, stream>>>(Q, K, V, O);
    mha_out_gemm<<<dim3(IN_ / 128, M_ / 128), 256, 0, stream>>>(O, w_out, b_out, out);
}
